// Codebook_4733053960562
// MI455X (gfx1250) — matrix operand order repaired and checked by reading, not compiled, not run on hardware
//
#include <hip/hip_runtime.h>

typedef __attribute__((ext_vector_type(16))) __bf16 v16bf;
typedef __attribute__((ext_vector_type(8)))  float  v8f;

#define EMBED 256
#define HW    1024
#define NTOT  32768
#define TILE  128

__device__ __forceinline__ unsigned short f2bf(float f) {
  unsigned u = __float_as_uint(f);
  u += 0x7FFFu + ((u >> 16) & 1u);          // round-to-nearest-even
  return (unsigned short)(u >> 16);
}

// ---------------------------------------------------------------------------
// Prep: bf16 codebook pre-swizzled into WMMA B-operand layout + squared norms
// wsw layout: [kchunk(64)][kstep(8)][lane(32)][8 dwords = 16 bf16]
//   B (32x16): lane = col + 16*(kk>=16), slot = kk&15  (contiguous k order, same as A staging)
// ---------------------------------------------------------------------------
__global__ void vq_prep(const float* __restrict__ w,
                        unsigned*    __restrict__ wswU32,
                        float*       __restrict__ wsq) {
  const int k = blockIdx.x;                  // codeword 0..1023
  const int t = threadIdx.x;                 // 0..127 -> d pair
  float2 a = ((const float2*)(w + (size_t)k * EMBED))[t];
  const int d0 = t << 1;
  const int kchunk = k >> 4, col = k & 15;
  const int kstep = d0 >> 5, kk = d0 & 31;
  const int lane = col + ((kk >> 4) << 4);
  const int pos  = kk & 15;                  // even; pair packs pos, pos+1
  wswU32[(((kchunk * 8 + kstep) * 32 + lane) << 3) + (pos >> 1)] =
      (unsigned)f2bf(a.x) | ((unsigned)f2bf(a.y) << 16);

  float s = a.x * a.x + a.y * a.y;
  #pragma unroll
  for (int off = 16; off > 0; off >>= 1) s += __shfl_down(s, off, 32);
  __shared__ float red[4];
  if ((t & 31) == 0) red[t >> 5] = s;
  __syncthreads();
  if (t == 0) wsq[k] = red[0] + red[1] + red[2] + red[3];
}

// ---------------------------------------------------------------------------
// Main: fused transpose-stage -> bf16 WMMA GEMM -> argmin -> all 4 outputs
// ---------------------------------------------------------------------------
__global__ void __launch_bounds__(256) vq_main(
    const float* __restrict__ x,       // [32][256][32][32]
    const float* __restrict__ w,       // [1024][256] fp32
    const v16bf* __restrict__ wswv,    // swizzled bf16 codebook
    const float* __restrict__ wsq,     // [1024]
    float* __restrict__ out) {
  __shared__ __align__(16) unsigned char shmem[65536];
  unsigned* zswU32 = (unsigned*)shmem;           // A-operand staging (64 KB)
  int*   tileIdx = (int*)shmem;                  // aliased after compute
  float* buf     = (float*)(shmem + 1024);       // aliased output staging

  const int t    = threadIdx.x;
  const int wv   = t >> 5;
  const int lane = t & 31;
  const int nbase = blockIdx.x * TILE;           // 256 tiles of 128 rows
  const int b   = nbase >> 10;
  const int hw0 = nbase & 1023;
  const float* xb = x + (size_t)b * (EMBED * HW) + hw0;

  float* out_ma = out;                                   // [B,D,H,W]
  float* out_zq = out + (size_t)NTOT * EMBED;            // [N,D]
  float* out_z  = out + (size_t)2 * NTOT * EMBED;        // [N,D]
  int*   out_ix = (int*)(out + (size_t)3 * NTOT * EMBED);// [N] int32

  // ---- stage z tile as bf16, pre-swizzled into WMMA A-operand layout ----
  // A (16x32): lane = row + 16*(kk>=16), slot = kk&15
  #pragma unroll 4
  for (int it = 0; it < 64; ++it) {
    const int p  = it * 2 + (t >> 7);            // d-pair 0..127
    const int hw = t & 127;
    const int d0 = p << 1;
    const float f0 = xb[d0 * HW + hw];
    const float f1 = xb[d0 * HW + HW + hw];
    const int rowtile = hw >> 4, row = hw & 15;
    const int kstep = d0 >> 5, kk = d0 & 31;
    const int alane = row + ((kk >> 4) << 4);
    const int pos   = kk & 15;                       // even
    zswU32[(((rowtile * 8 + kstep) * 32 + alane) << 3) + (pos >> 1)] =
        (unsigned)f2bf(f0) | ((unsigned)f2bf(f1) << 16);
  }
  __syncthreads();

  // ---- per-wave GEMM over K=1024 with running argmin ----
  const v16bf* Az = (const v16bf*)shmem;
  v16bf a[8];
  #pragma unroll
  for (int ks = 0; ks < 8; ++ks) a[ks] = Az[(wv * 8 + ks) * 32 + lane];

  float best[8];
  int   bidx[8];
  #pragma unroll
  for (int r = 0; r < 8; ++r) { best[r] = 3.4e38f; bidx[r] = 0; }

  const char* pfbase = (const char*)wswv + lane * 256;   // 32 lanes x 256B = 8KB chunk

  #pragma unroll 2
  for (int kc = 0; kc < 64; ++kc) {
    // pull next chunk's 8KB B-region toward WGP$ while this chunk computes
    if (kc + 1 < 64)
      __builtin_prefetch(pfbase + (size_t)(kc + 1) * 8192, 0, 3);

    // issue the whole chunk's B loads as one clause, then WMMA with
    // incremental loadcnt waits instead of per-pair full stalls
    v16bf bm[8];
    #pragma unroll
    for (int ks = 0; ks < 8; ++ks)
      bm[ks] = wswv[(kc * 8 + ks) * 32 + lane];

    const int kcol = (kc << 4) + (lane & 15);
    const float wq = wsq[kcol];

    v8f c = {0.f, 0.f, 0.f, 0.f, 0.f, 0.f, 0.f, 0.f};
    #pragma unroll
    for (int ks = 0; ks < 8; ++ks)
      c = __builtin_amdgcn_wmma_f32_16x16x32_bf16(
              false, a[ks], false, bm[ks], (short)0, c, false, false);

    #pragma unroll
    for (int r = 0; r < 8; ++r) {
      float s = fmaf(-2.f, c[r], wq);            // + z_sq is row-constant
      if (s < best[r]) { best[r] = s; bidx[r] = kcol; }
    }
  }

  // ---- argmin across the 16 lanes holding each C row (first-index ties) ----
  int myIdx = 0;
  #pragma unroll
  for (int r = 0; r < 8; ++r) {
    float bv = best[r]; int bi = bidx[r];
    #pragma unroll
    for (int m = 8; m > 0; m >>= 1) {            // stays inside 16-lane half
      float ov = __shfl_xor(bv, m, 32);
      int   oi = __shfl_xor(bi, m, 32);
      if (ov < bv || (ov == bv && oi < bi)) { bv = ov; bi = oi; }
    }
    if ((lane & 15) == r) myIdx = bi;            // lane r -> row r; r+16 -> row r+8
  }
  __syncthreads();                               // zsw reads done; alias tileIdx
  if ((lane & 15) < 8) {
    const int rowInTile = wv * 16 + (lane & 15) + ((lane >> 4) << 3);
    tileIdx[rowInTile] = myIdx;
    out_ix[nbase + rowInTile] = myIdx;
  }
  __syncthreads();

  // ---- z_q (coalesced gather) + z_q_ma (transposed via padded LDS) ----
  for (int sub = 0; sub < 4; ++sub) {
    for (int rr = 0; rr < 32; ++rr) {
      const int n  = nbase + sub * 32 + rr;
      const int ci = tileIdx[sub * 32 + rr];
      const float v = w[(size_t)ci * EMBED + t];
      out_zq[(size_t)n * EMBED + t] = v;
      buf[rr * 261 + t] = v;                     // stride%64==5: conflict-free
    }
    __syncthreads();
    const int rl = t & 31, dl = t >> 5;
    for (int db = 0; db < 32; ++db) {
      const int d = db * 8 + dl;
      out_ma[(size_t)b * (EMBED * HW) + d * HW + hw0 + sub * 32 + rl] =
          buf[rl * 261 + d];
    }
    __syncthreads();
  }

  // ---- z = transpose of x tile via padded LDS ----
  float* zb = buf;
  for (int dsub = 0; dsub < 8; ++dsub) {
    for (int j = 0; j < 16; ++j) {
      const int dloc = j * 2 + (t >> 7);
      const int hw = t & 127;
      zb[dloc * 133 + hw] = xb[(dsub * 32 + dloc) * HW + hw];
    }
    __syncthreads();
    for (int j = 0; j < 16; ++j) {
      const int hw = j * 8 + (t >> 5);
      const int dloc = t & 31;
      out_z[(size_t)(nbase + hw) * EMBED + dsub * 32 + dloc] =
          zb[dloc * 133 + hw];
    }
    __syncthreads();
  }
}

extern "C" void kernel_launch(void* const* d_in, const int* in_sizes, int n_in,
                              void* d_out, int out_size, void* d_ws, size_t ws_size,
                              hipStream_t stream) {
  const float* x = (const float*)d_in[0];        // 32*256*32*32
  const float* w = (const float*)d_in[1];        // 1024*256
  float* out = (float*)d_out;
  // workspace: 512 KB swizzled bf16 codebook + 4 KB norms
  unsigned* wsw = (unsigned*)d_ws;
  float*    wsq = (float*)((char*)d_ws + 524288);

  vq_prep<<<1024, 128, 0, stream>>>(w, wsw, wsq);
  vq_main<<<256, 256, 0, stream>>>(x, w, (const v16bf*)wsw, wsq, out);
}
